// Attention_23725399343190
// MI455X (gfx1250) — compile-verified, matrix operand order checked by reading, not run on hardware
//
#include <hip/hip_runtime.h>
#include <hip/hip_bf16.h>

typedef __attribute__((ext_vector_type(16))) __bf16 v16bf;
typedef __attribute__((ext_vector_type(8)))  __bf16 v8bf;
typedef __attribute__((ext_vector_type(8)))  float  v8f;

__device__ __forceinline__ v8f wmma_bf16(v16bf a, v16bf b, v8f c) {
  // v_wmma_f32_16x16x32_bf16: D = A(16x32) * B(32x16) + C(16x16 f32)
  return __builtin_amdgcn_wmma_f32_16x16x32_bf16(
      /*neg_a=*/false, a, /*neg_b=*/false, b,
      /*c_mod=*/(short)0, c, /*reuse_a=*/false, /*reuse_b=*/false);
}

// Assemble a 16-element fragment from two contiguous 8-element halves.
// Per ISA 7.12.2 (16-bit A/B layout): for lane-half kh, fragment element i<8
// corresponds to k = kh*8 + i, element i>=8 to k = 16 + kh*8 + (i-8).
// So lo = &data[kh*8], hi = &data[16 + kh*8] (both 16B-aligned contiguous).
__device__ __forceinline__ v16bf mk_frag(v8bf lo, v8bf hi) {
  v16bf f;
#pragma unroll
  for (int i = 0; i < 8; ++i) { f[i] = lo[i]; f[i + 8] = hi[i]; }
  return f;
}

// ---------------------------------------------------------------------------
// Generic WMMA GEMM: C[M,N] = A[M,K] @ B[K,N] (+bias), f32 inputs staged to
// bf16 in LDS. Block tile 128x128, K-step 32, register-prefetch pipeline.
// 256 threads = 8 waves (4x2 grid); each wave computes 32x64 as 2x4 WMMAs.
// ---------------------------------------------------------------------------
template <bool OUT_BF16, bool ADD_BIAS>
__global__ __launch_bounds__(256) void gemm_wmma_kernel(
    const float* __restrict__ A, const float* __restrict__ B,
    const float* __restrict__ bias, void* __restrict__ Cout,
    int M, int N, int K) {
  constexpr int LDT = 40;  // padded bf16 stride (80B, keeps 16B alignment)
  __shared__ __bf16 As[128 * LDT];  // 128 rows x 32 k
  __shared__ __bf16 Bt[128 * LDT];  // 128 cols x 32 k (B transposed)

  const int tid  = threadIdx.x;
  const int lane = tid & 31;
  const int wave = tid >> 5;
  const int bm = blockIdx.y * 128;
  const int bn = blockIdx.x * 128;
  const int wr = wave >> 1;  // 0..3 -> 32-row block
  const int wc = wave & 1;   // 0..1 -> 64-col block
  const int kh  = lane >> 4;
  const int l15 = lane & 15;

  // Staging coordinates (fixed per thread).
  const int ar  = tid >> 1, ac0 = (tid & 1) * 16;   // A: 2 thr/row, 16 f32 each
  const int bk  = tid & 31, bn0 = (tid >> 5) * 16;  // B: 16 f32 per thread
  const float* aSrc = A + (size_t)(bm + ar) * K + ac0;
  const float* bSrc = B + (size_t)bk * N + bn + bn0;

  // Prologue: prefetch first K-chunk into registers.
  float aR[16], bR[16];
#pragma unroll
  for (int i = 0; i < 16; ++i) aR[i] = aSrc[i];
#pragma unroll
  for (int i = 0; i < 16; ++i) bR[i] = bSrc[i];

  v8f acc[2][4] = {};

  for (int k0 = 0; k0 < K; k0 += 32) {
    // Commit prefetched chunk to LDS (f32 -> bf16).
#pragma unroll
    for (int i = 0; i < 16; ++i) As[ar * LDT + ac0 + i] = (__bf16)aR[i];
#pragma unroll
    for (int i = 0; i < 16; ++i) Bt[(bn0 + i) * LDT + bk] = (__bf16)bR[i];
    __syncthreads();

    // Prefetch next chunk; global loads overlap the WMMAs below.
    if (k0 + 32 < K) {
      const float* aN = aSrc + (k0 + 32);
      const float* bN = bSrc + (size_t)(k0 + 32) * N;
#pragma unroll
      for (int i = 0; i < 16; ++i) aR[i] = aN[i];
#pragma unroll
      for (int i = 0; i < 16; ++i) bR[i] = bN[i];
    }

    v16bf af[2], bfr[4];
#pragma unroll
    for (int rt = 0; rt < 2; ++rt) {
      const int row = wr * 32 + rt * 16 + l15;
      const v8bf* p0 = (const v8bf*)&As[row * LDT + kh * 8];
      const v8bf* p1 = (const v8bf*)&As[row * LDT + 16 + kh * 8];
      af[rt] = mk_frag(*p0, *p1);
    }
#pragma unroll
    for (int ct = 0; ct < 4; ++ct) {
      const int col = wc * 64 + ct * 16 + l15;
      const v8bf* p0 = (const v8bf*)&Bt[col * LDT + kh * 8];
      const v8bf* p1 = (const v8bf*)&Bt[col * LDT + 16 + kh * 8];
      bfr[ct] = mk_frag(*p0, *p1);
    }
#pragma unroll
    for (int rt = 0; rt < 2; ++rt)
#pragma unroll
      for (int ct = 0; ct < 4; ++ct)
        acc[rt][ct] = wmma_bf16(af[rt], bfr[ct], acc[rt][ct]);
    __syncthreads();
  }

  // Epilogue: C/D layout -> element r: row = r + kh*8, col = l15.
#pragma unroll
  for (int rt = 0; rt < 2; ++rt) {
#pragma unroll
    for (int ct = 0; ct < 4; ++ct) {
      const int col = bn + wc * 64 + ct * 16 + l15;
#pragma unroll
      for (int r = 0; r < 8; ++r) {
        const int row = bm + wr * 32 + rt * 16 + r + kh * 8;
        float v = acc[rt][ct][r];
        if (ADD_BIAS) v += bias[col];
        if (OUT_BF16)
          ((__bf16*)Cout)[(size_t)row * N + col] = (__bf16)v;
        else
          ((float*)Cout)[(size_t)row * N + col] = v;
      }
    }
  }
}

// ---------------------------------------------------------------------------
// Fused attention: per (b, h, 32-query tile):
//   Phase 1: S = (Q K^T) * hd^-0.5 into 128 KB LDS tile (WMMA bf16)
//   Phase 2: exact row softmax, write normalized P (float4) to attn output
//   Phase 3: ctx = P @ V; V tiles staged transposed through LDS so both the
//            global loads (b128, coalesced) and fragment reads (ds b128) are
//            wide, instead of 2-byte strided gathers.
// qkv buffer: bf16, layout [B, N, 3C] with col = s*1024 + h*64 + d.
// ---------------------------------------------------------------------------
__global__ __launch_bounds__(256) void attn_kernel(
    const __bf16* __restrict__ qkv, float* __restrict__ attn_out,
    float* __restrict__ ctx) {
  constexpr int N = 1024, C3 = 3072, HD = 64;
  constexpr int LDV = 40;
  __shared__ float S[32 * N];        // 128 KB score/probability tile
  __shared__ __bf16 Vt[64 * LDV];    // 5 KB transposed V tile [d][key]
  __shared__ float redmax[32 * 8];
  __shared__ float redsum[32 * 8];

  const int wg = blockIdx.x;
  const int qt = wg & 31;            // query tile (32 rows)
  const int bh = wg >> 5;            // b*16 + h
  const int b = bh >> 4, h = bh & 15;
  const int tid = threadIdx.x, lane = tid & 31, wave = tid >> 5;
  const int kh = lane >> 4, l15 = lane & 15;
  const size_t base = (size_t)b * N * C3;
  const int qcol = h * HD;
  const int kcol = 1024 + h * HD;
  const int vcol = 2048 + h * HD;

  // ---- Phase 1: S = Q K^T * scale ------------------------------------
  v16bf qf[2][2];  // [row tile][k chunk of 32 within hd=64]
#pragma unroll
  for (int rt = 0; rt < 2; ++rt) {
    const int qrow = qt * 32 + rt * 16 + l15;
    const __bf16* qp = qkv + base + (size_t)qrow * C3 + qcol;
#pragma unroll
    for (int kc = 0; kc < 2; ++kc) {
      const v8bf* p0 = (const v8bf*)(qp + kc * 32 + kh * 8);
      const v8bf* p1 = (const v8bf*)(qp + kc * 32 + 16 + kh * 8);
      qf[rt][kc] = mk_frag(*p0, *p1);
    }
  }
  const int key0 = wave * 128;  // 8 waves cover 1024 keys
  for (int ci = 0; ci < 8; ++ci) {
    const int key = key0 + ci * 16 + l15;  // B-frag lane = key column
    const __bf16* kp = qkv + base + (size_t)key * C3 + kcol;
    v8f acc0 = {}, acc1 = {};
#pragma unroll
    for (int kc = 0; kc < 2; ++kc) {
      const v8bf* p0 = (const v8bf*)(kp + kc * 32 + kh * 8);
      const v8bf* p1 = (const v8bf*)(kp + kc * 32 + 16 + kh * 8);
      const v16bf kf = mk_frag(*p0, *p1);
      acc0 = wmma_bf16(qf[0][kc], kf, acc0);
      acc1 = wmma_bf16(qf[1][kc], kf, acc1);
    }
    const float scale = 0.125f;  // 64^-0.5
    const int colk = key0 + ci * 16 + l15;
#pragma unroll
    for (int r = 0; r < 8; ++r) {
      S[(r + kh * 8) * N + colk]      = acc0[r] * scale;
      S[(16 + r + kh * 8) * N + colk] = acc1[r] * scale;
    }
  }
  __syncthreads();

  // ---- Phase 2: row softmax + write attn (float4) ----------------------
  {
    const int r = tid & 31, seg = tid >> 5;  // 8 segments of 128 per row
    float4* Srow = (float4*)&S[r * N + seg * 128];
    float m = -3.4e38f;
    for (int i = 0; i < 32; ++i) {
      const float4 v = Srow[i];
      m = fmaxf(m, fmaxf(fmaxf(v.x, v.y), fmaxf(v.z, v.w)));
    }
    redmax[r * 8 + seg] = m;
    __syncthreads();
    float rowmax = redmax[r * 8];
#pragma unroll
    for (int i = 1; i < 8; ++i) rowmax = fmaxf(rowmax, redmax[r * 8 + i]);
    float s = 0.f;
    for (int i = 0; i < 32; ++i) {
      float4 v = Srow[i];
      v.x = __expf(v.x - rowmax); v.y = __expf(v.y - rowmax);
      v.z = __expf(v.z - rowmax); v.w = __expf(v.w - rowmax);
      Srow[i] = v;
      s += (v.x + v.y) + (v.z + v.w);
    }
    redsum[r * 8 + seg] = s;
    __syncthreads();
    float rowsum = 0.f;
#pragma unroll
    for (int i = 0; i < 8; ++i) rowsum += redsum[r * 8 + i];
    const float inv = 1.0f / rowsum;
    float4* aout = (float4*)(attn_out + ((size_t)bh * N + qt * 32 + r) * N +
                             seg * 128);
    for (int i = 0; i < 32; ++i) {
      float4 v = Srow[i];
      v.x *= inv; v.y *= inv; v.z *= inv; v.w *= inv;
      Srow[i] = v;   // keep normalized P in LDS for phase 3
      aout[i] = v;   // global_store_b128
    }
  }

  // ---- Phase 3: ctx = P @ V (32x64 output, one 16x16 tile per wave) ---
  {
    const int rt = wave >> 2;  // 0..1 row tile
    const int ct = wave & 3;   // 0..3 col tile (16 d-columns)
    const int skey = tid >> 3;           // staging: key within tile (0..31)
    const int sd0  = (tid & 7) * 8;      // staging: 8 d-values per thread
    v8f acc = {};
    for (int kc = 0; kc < 32; ++kc) {  // K = 1024 keys, 32 per step
      // Coalesced b128 load of V[kc*32+skey][sd0..sd0+8) ...
      const v8bf vv = *(const v8bf*)(qkv + base +
                                     (size_t)(kc * 32 + skey) * C3 + vcol + sd0);
      __syncthreads();  // previous iteration's Vt readers are done
      // ... stored transposed: Vt[d][key]
#pragma unroll
      for (int i = 0; i < 8; ++i) Vt[(sd0 + i) * LDV + skey] = vv[i];
      __syncthreads();

      // A frag from P in LDS (f32 -> bf16)
      const int prow = rt * 16 + l15;
      const float* ps = &S[prow * N + kc * 32];
      v16bf af;
#pragma unroll
      for (int i = 0; i < 8; ++i) {
        af[i]     = (__bf16)ps[kh * 8 + i];
        af[i + 8] = (__bf16)ps[16 + kh * 8 + i];
      }
      // B frag from transposed V tile: lane = d column, contiguous in key
      const int d = ct * 16 + l15;
      const v8bf* q0 = (const v8bf*)&Vt[d * LDV + kh * 8];
      const v8bf* q1 = (const v8bf*)&Vt[d * LDV + 16 + kh * 8];
      acc = wmma_bf16(af, mk_frag(*q0, *q1), acc);
    }
#pragma unroll
    for (int r = 0; r < 8; ++r) {
      const int n = qt * 32 + rt * 16 + r + kh * 8;
      const int c = h * HD + ct * 16 + l15;
      ctx[((size_t)b * N + n) * 1024 + c] = acc[r];
    }
  }
}

// ---------------------------------------------------------------------------
extern "C" void kernel_launch(void* const* d_in, const int* in_sizes, int n_in,
                              void* d_out, int out_size, void* d_ws,
                              size_t ws_size, hipStream_t stream) {
  (void)in_sizes; (void)n_in; (void)out_size; (void)ws_size;
  const float* x      = (const float*)d_in[0];  // [8,1024,1024]
  const float* w_qkv  = (const float*)d_in[1];  // [1024,3072]
  const float* w_proj = (const float*)d_in[2];  // [1024,1024]
  const float* b_proj = (const float*)d_in[3];  // [1024]

  float* out  = (float*)d_out;                        // [8,1024,1024]
  float* attn = out + (size_t)8 * 1024 * 1024;        // [8,16,1024,1024]

  __bf16* qkv = (__bf16*)d_ws;                                        // 48 MB
  float*  ctx = (float*)((char*)d_ws + (size_t)8 * 1024 * 3072 * 2);  // 32 MB

  // 1) qkv = x @ w_qkv  (M=8192, N=3072, K=1024), bf16 output
  gemm_wmma_kernel<true, false>
      <<<dim3(3072 / 128, 8192 / 128), 256, 0, stream>>>(
          x, w_qkv, nullptr, qkv, 8192, 3072, 1024);

  // 2) fused QK^T -> softmax -> attn write -> PV
  attn_kernel<<<dim3(8 * 16 * 32), 256, 0, stream>>>(qkv, attn, ctx);

  // 3) out = ctx @ w_proj + b_proj  (M=8192, N=1024, K=1024)
  gemm_wmma_kernel<false, true>
      <<<dim3(1024 / 128, 8192 / 128), 256, 0, stream>>>(
          ctx, w_proj, b_proj, out, 8192, 1024, 1024);
}
